// TransformerBlock_31275951850097
// MI455X (gfx1250) — matrix operand order repaired and checked by reading, not compiled, not run on hardware
//
#include <hip/hip_runtime.h>

typedef __attribute__((ext_vector_type(16))) __bf16 v16bf;
typedef __attribute__((ext_vector_type(8)))  __bf16 v8bf;
typedef __attribute__((ext_vector_type(8)))  float  v8f;

// ---------------------------------------------------------------------------
// WMMA helpers (CDNA5 bf16, 16x16x32, f32 accum)
// ---------------------------------------------------------------------------
__device__ __forceinline__ v8f wmma_bf16(v16bf a, v16bf b, v8f c) {
  return __builtin_amdgcn_wmma_f32_16x16x32_bf16(
      /*neg_a=*/false, a, /*neg_b=*/false, b,
      /*c_mod=*/(short)0, c, /*reuse_a=*/false, /*reuse_b=*/false);
}

// A-fragment (16x32 bf16): lane group g=lane/16 holds row M=lane%16,
// K chunk {g*16 .. g*16+15} (contiguous, matching B fragments).
__device__ __forceinline__ v16bf load_a_frag(const __bf16* p) {
  v8bf lo = *(const v8bf*)(p);
  v8bf hi = *(const v8bf*)(p + 8);
  v16bf a;
#pragma unroll
  for (int i = 0; i < 8; ++i) { a[i] = lo[i]; a[i + 8] = hi[i]; }
  return a;
}

// ---------------------------------------------------------------------------
// Transposed f32 -> bf16 weight conversion: out[N][K] = (bf16) in[K][N].
// ---------------------------------------------------------------------------
__global__ __launch_bounds__(256) void transpose_to_bf16_kernel(
    const float* __restrict__ in, __bf16* __restrict__ out, int K, int N) {
  __shared__ float tile[32][33];
  const int kb = blockIdx.y * 32, nb = blockIdx.x * 32;
  const int tx = threadIdx.x & 31, ty = threadIdx.x >> 5;  // 32 x 8
#pragma unroll
  for (int i = 0; i < 32; i += 8)
    tile[ty + i][tx] = in[(size_t)(kb + ty + i) * N + nb + tx];
  __syncthreads();
#pragma unroll
  for (int i = 0; i < 32; i += 8)
    out[(size_t)(nb + ty + i) * K + kb + tx] = (__bf16)tile[tx][ty + i];
}

// ---------------------------------------------------------------------------
// RMSNorm: one block per row of D=1024, float4 loads, packed bf16x4 stores
// ---------------------------------------------------------------------------
__global__ __launch_bounds__(256) void rmsnorm_bf16_kernel(
    const float* __restrict__ x, const float* __restrict__ g,
    __bf16* __restrict__ out, int D) {
  __shared__ float red[8];
  const int row = blockIdx.x;
  const float4* xr = (const float4*)(x + (size_t)row * D);
  const float4* g4 = (const float4*)g;
  const int n4 = D >> 2;
  float ss = 0.f;
  for (int i = threadIdx.x; i < n4; i += blockDim.x) {
    float4 t = xr[i];
    ss += t.x * t.x + t.y * t.y + t.z * t.z + t.w * t.w;
  }
  ss += __shfl_xor(ss, 1);
  ss += __shfl_xor(ss, 2);
  ss += __shfl_xor(ss, 4);
  ss += __shfl_xor(ss, 8);
  ss += __shfl_xor(ss, 16);
  if ((threadIdx.x & 31) == 0) red[threadIdx.x >> 5] = ss;
  __syncthreads();
  float tot = 0.f;
#pragma unroll
  for (int i = 0; i < 8; ++i) tot += red[i];
  const float norm = rsqrtf(tot / (float)D + 1e-5f);
  uint2* orow = (uint2*)(out + (size_t)row * D);
  for (int i = threadIdx.x; i < n4; i += blockDim.x) {
    float4 t = xr[i];
    float4 gg = g4[i];
    union { __bf16 h[4]; uint2 u; } o;
    o.h[0] = (__bf16)(t.x * norm * gg.x);
    o.h[1] = (__bf16)(t.y * norm * gg.y);
    o.h[2] = (__bf16)(t.z * norm * gg.z);
    o.h[3] = (__bf16)(t.w * norm * gg.w);
    orow[i] = o.u;
  }
}

// ---------------------------------------------------------------------------
// Tiled WMMA GEMM: C[M,N] = act( A[M,K] @ Bt[N,K]^T + bias ) + resid
// Block tile 128x128, 4 waves, wave tile 64x64 (4x4 WMMA tiles), K-step 32.
// Double-buffered LDS, software-pipelined (loads before WMMA, stores after).
// ---------------------------------------------------------------------------
enum { OUT_F32 = 0, OUT_BF16 = 1, OUT_QKV = 2 };

template <bool BIAS, bool SILU, bool RES, int OUTM>
__global__ __launch_bounds__(128) void gemm_bf16_wmma_kernel(
    const __bf16* __restrict__ A, const __bf16* __restrict__ Bt,
    const float* __restrict__ bias, const float* __restrict__ resid,
    float* __restrict__ outF, __bf16* __restrict__ outB,
    __bf16* __restrict__ outQ, __bf16* __restrict__ outK,
    __bf16* __restrict__ outVT, int M, int N, int K) {
  __shared__ __align__(64) __bf16 ldsA[2][128][32];
  __shared__ __align__(64) __bf16 ldsB[2][128][32];

  const int tid  = threadIdx.x;
  const int lane = tid & 31;
  const int w    = tid >> 5;   // 0..3
  const int wr   = w & 1;      // wave row (64 rows)
  const int wc   = w >> 1;     // wave col (64 cols)
  const int g16  = lane >> 4;
  const int ln   = lane & 15;
  const int m0   = blockIdx.y * 128;
  const int n0   = blockIdx.x * 128;

  const __bf16* gA = A  + (size_t)(m0 + tid) * K;  // one full 32-wide row/thread
  const __bf16* gB = Bt + (size_t)(n0 + tid) * K;

  v8f zero = {};
  v8f acc[4][4];
#pragma unroll
  for (int mi = 0; mi < 4; ++mi)
#pragma unroll
    for (int ni = 0; ni < 4; ++ni) acc[mi][ni] = zero;

  // ---- stage tile 0 ----
  {
    const uint4* pa = (const uint4*)gA;
    const uint4* pb = (const uint4*)gB;
    uint4 a0 = pa[0], a1 = pa[1], a2 = pa[2], a3 = pa[3];
    uint4 b0 = pb[0], b1 = pb[1], b2 = pb[2], b3 = pb[3];
    uint4* la = (uint4*)&ldsA[0][tid][0];
    la[0] = a0; la[1] = a1; la[2] = a2; la[3] = a3;
    uint4* lb = (uint4*)&ldsB[0][tid][0];
    lb[0] = b0; lb[1] = b1; lb[2] = b2; lb[3] = b3;
  }
  __syncthreads();

  const int nk = K / 32;
  for (int t = 0; t < nk; ++t) {
    const int buf = t & 1;
    const bool more = (t + 1 < nk);
    uint4 a0{}, a1{}, a2{}, a3{}, b0{}, b1{}, b2{}, b3{};
    if (more) {  // next-tile global loads before the WMMA block
      const uint4* pa = (const uint4*)(gA + (size_t)(t + 1) * 32);
      const uint4* pb = (const uint4*)(gB + (size_t)(t + 1) * 32);
      a0 = pa[0]; a1 = pa[1]; a2 = pa[2]; a3 = pa[3];
      b0 = pb[0]; b1 = pb[1]; b2 = pb[2]; b3 = pb[3];
      if (t + 2 < nk) {
        __builtin_prefetch(gA + (size_t)(t + 2) * 32, 0, 1);
        __builtin_prefetch(gB + (size_t)(t + 2) * 32, 0, 1);
      }
    }

    v16bf bfr[4];
#pragma unroll
    for (int ni = 0; ni < 4; ++ni)
      bfr[ni] = *(const v16bf*)&ldsB[buf][wc * 64 + ni * 16 + ln][g16 * 16];
#pragma unroll
    for (int mi = 0; mi < 4; ++mi) {
      v16bf af = load_a_frag(&ldsA[buf][wr * 64 + mi * 16 + ln][g16 * 16]);
#pragma unroll
      for (int ni = 0; ni < 4; ++ni)
        acc[mi][ni] = wmma_bf16(af, bfr[ni], acc[mi][ni]);
    }

    if (more) {  // LDS stores after compute
      uint4* la = (uint4*)&ldsA[buf ^ 1][tid][0];
      la[0] = a0; la[1] = a1; la[2] = a2; la[3] = a3;
      uint4* lb = (uint4*)&ldsB[buf ^ 1][tid][0];
      lb[0] = b0; lb[1] = b1; lb[2] = b2; lb[3] = b3;
    }
    __syncthreads();
  }

  // ---- epilogue: C layout VGPR r -> M = r + 8*g16, N = lane%16 ----
  const int seg = (OUTM == OUT_QKV) ? (n0 >> 10) : 0;  // block-uniform
#pragma unroll
  for (int mi = 0; mi < 4; ++mi)
#pragma unroll
    for (int ni = 0; ni < 4; ++ni)
#pragma unroll
      for (int r = 0; r < 8; ++r) {
        const int row = m0 + wr * 64 + mi * 16 + r + 8 * g16;
        const int col = n0 + wc * 64 + ni * 16 + ln;
        float v = acc[mi][ni][r];
        if (BIAS) v += bias[col];
        if (SILU) v = v * __builtin_amdgcn_rcpf(1.f + __expf(-v));
        if (RES) v += resid[(size_t)row * N + col];
        if (OUTM == OUT_F32) {
          outF[(size_t)row * N + col] = v;
        } else if (OUTM == OUT_BF16) {
          outB[(size_t)row * N + col] = (__bf16)v;
        } else {  // OUT_QKV: N==3072, D==1024, T==2048
          const int colL = col & 1023;
          if (seg == 0) {
            outQ[(size_t)row * 1024 + colL] = (__bf16)(v * 0.125f);
          } else if (seg == 1) {
            outK[(size_t)row * 1024 + colL] = (__bf16)v;
          } else {
            const int bb = row >> 11;
            const int tt = row & 2047;
            outVT[((size_t)(bb * 1024 + colL) << 11) + tt] = (__bf16)v;
          }
        }
      }
}

// ---------------------------------------------------------------------------
// Flash attention (causal, HD=64): one WG = one (b,h) x 128 query rows.
// 8 waves x 16 q rows. q pre-scaled by 1/sqrt(HD) at projection time.
// Per-tile wave-uniform classification (scalar branches via readfirstlane):
//   kmax <= qmin : below diagonal  -> no masking, no WMMA->VALU hazard NOPs
//   kmin >  qmax : above diagonal  -> tile contributes zero, skip compute
//   otherwise    : diagonal tile   -> masked path
// ---------------------------------------------------------------------------
__global__ __launch_bounds__(256) void flash_attn_bf16_kernel(
    const __bf16* __restrict__ Q, const __bf16* __restrict__ Kp,
    const __bf16* __restrict__ VT, __bf16* __restrict__ O) {
  constexpr int T = 2048, D = 1024, HD = 64, H = 16;
  __shared__ __align__(64) __bf16 ldsQ[128][64];
  __shared__ __align__(64) __bf16 ldsK[2][64][64];   // [key][d]
  __shared__ __align__(64) __bf16 ldsVt[2][64][64];  // [d][key]
  __shared__ __align__(64) __bf16 ldsP[8][16][64];   // per-wave P tile

  const int tid  = threadIdx.x;
  const int lane = tid & 31;
  const int w    = tid >> 5;
  const int g16  = lane >> 4;
  const int ln   = lane & 15;

  const int nqt   = T / 128;
  const int bh    = blockIdx.x / nqt;
  const int qt    = blockIdx.x % nqt;
  const int b     = bh / H;
  const int h     = bh % H;
  const int qbase = qt * 128;
  const size_t rowoff = (size_t)(b * T) * D + h * HD;

  // wave-uniform query-row bounds of this wave's 16-row strip
  const int wu   = __builtin_amdgcn_readfirstlane(w);
  const int qmin = qbase + wu * 16;
  const int qmax = qmin + 15;

  const int kk  = tid >> 2;
  const int seg = tid & 3;
  const __bf16* gK = Kp + rowoff + (size_t)kk * D + seg * 16;
  const __bf16* gV = VT + ((size_t)(b * 1024 + h * HD + kk) << 11) + seg * 16;

  // ---- stage Q (128x64) and K/V tile 0 ----
  {
    const int r = tid >> 1, half = tid & 1;
    const uint4* gq =
        (const uint4*)(Q + rowoff + (size_t)(qbase + r) * D + half * 32);
    uint4* lq = (uint4*)&ldsQ[r][half * 32];
    lq[0] = gq[0]; lq[1] = gq[1]; lq[2] = gq[2]; lq[3] = gq[3];

    const uint4* pk = (const uint4*)gK;
    const uint4* pv = (const uint4*)gV;
    uint4 k0 = pk[0], k1 = pk[1], v0 = pv[0], v1 = pv[1];
    uint4* lk = (uint4*)&ldsK[0][kk][seg * 16];
    lk[0] = k0; lk[1] = k1;
    uint4* lv = (uint4*)&ldsVt[0][kk][seg * 16];
    lv[0] = v0; lv[1] = v1;
  }
  __syncthreads();

  v16bf qa[2];
#pragma unroll
  for (int s = 0; s < 2; ++s)
    qa[s] = load_a_frag(&ldsQ[w * 16 + ln][s * 32 + g16 * 16]);

  v8f zero = {};
  v8f ofrag[4];
#pragma unroll
  for (int di = 0; di < 4; ++di) ofrag[di] = zero;
  float m_i[8], l_i[8];
#pragma unroll
  for (int r = 0; r < 8; ++r) { m_i[r] = -1e30f; l_i[r] = 0.f; }

  const int nkt = qbase / 64 + 2;  // causal bound: keys <= qbase+127
  for (int kt = 0; kt < nkt; ++kt) {
    const int buf = kt & 1;
    const bool more = (kt + 1 < nkt);
    uint4 k0{}, k1{}, v0{}, v1{};
    if (more) {
      const uint4* pk = (const uint4*)(gK + (size_t)(kt + 1) * 64 * D);
      const uint4* pv = (const uint4*)(gV + (size_t)(kt + 1) * 64);
      k0 = pk[0]; k1 = pk[1]; v0 = pv[0]; v1 = pv[1];
    }

    const int kmin = kt * 64, kmax = kmin + 63;
    if (kmin <= qmax) {  // else: fully above diagonal, contributes zero
      // ---- S = Q @ K^T (pre-scaled) ----
      v8f sfrag[4];
#pragma unroll
      for (int ni = 0; ni < 4; ++ni) {
        v8f s = zero;
#pragma unroll
        for (int ds = 0; ds < 2; ++ds) {
          v16bf bk =
              *(const v16bf*)&ldsK[buf][ni * 16 + ln][ds * 32 + g16 * 16];
          s = wmma_bf16(qa[ds], bk, s);
        }
        sfrag[ni] = s;
      }

      // ---- causal mask: only the diagonal tile needs it ----
      if (kmax > qmin) {
        const int qrow_base = qbase + w * 16 + 8 * g16;
#pragma unroll
        for (int ni = 0; ni < 4; ++ni)
#pragma unroll
          for (int r = 0; r < 8; ++r) {
            const int key = kmin + ni * 16 + ln;
            sfrag[ni][r] = (key <= qrow_base + r) ? sfrag[ni][r] : -1e9f;
          }
      }

      // ---- online softmax (row reductions across 16-lane N-groups) ----
      float scale_r[8];
#pragma unroll
      for (int r = 0; r < 8; ++r) {
        float mx = fmaxf(fmaxf(sfrag[0][r], sfrag[1][r]),
                         fmaxf(sfrag[2][r], sfrag[3][r]));
        mx = fmaxf(mx, __shfl_xor(mx, 1));
        mx = fmaxf(mx, __shfl_xor(mx, 2));
        mx = fmaxf(mx, __shfl_xor(mx, 4));
        mx = fmaxf(mx, __shfl_xor(mx, 8));
        const float mnew = fmaxf(m_i[r], mx);
        const float sc = __expf(m_i[r] - mnew);
        float ls = 0.f;
#pragma unroll
        for (int ni = 0; ni < 4; ++ni) {
          const float p = __expf(sfrag[ni][r] - mnew);
          sfrag[ni][r] = p;
          ls += p;
        }
        ls += __shfl_xor(ls, 1);
        ls += __shfl_xor(ls, 2);
        ls += __shfl_xor(ls, 4);
        ls += __shfl_xor(ls, 8);
        l_i[r] = l_i[r] * sc + ls;
        m_i[r] = mnew;
        scale_r[r] = sc;
      }
#pragma unroll
      for (int di = 0; di < 4; ++di)
#pragma unroll
        for (int r = 0; r < 8; ++r) ofrag[di][r] *= scale_r[r];

      // ---- write P (bf16) to wave-private LDS, re-read in A layout ----
#pragma unroll
      for (int ni = 0; ni < 4; ++ni)
#pragma unroll
        for (int r = 0; r < 8; ++r)
          ldsP[w][r + 8 * g16][ni * 16 + ln] = (__bf16)sfrag[ni][r];

      // ---- O += P @ V ----
#pragma unroll
      for (int ks = 0; ks < 2; ++ks) {
        v16bf pa = load_a_frag(&ldsP[w][ln][ks * 32 + g16 * 16]);
#pragma unroll
        for (int di = 0; di < 4; ++di) {
          v16bf vb =
              *(const v16bf*)&ldsVt[buf][di * 16 + ln][ks * 32 + g16 * 16];
          ofrag[di] = wmma_bf16(pa, vb, ofrag[di]);
        }
      }
    }

    if (more) {
      uint4* lk = (uint4*)&ldsK[buf ^ 1][kk][seg * 16];
      lk[0] = k0; lk[1] = k1;
      uint4* lv = (uint4*)&ldsVt[buf ^ 1][kk][seg * 16];
      lv[0] = v0; lv[1] = v1;
    }
    __syncthreads();
  }

  // ---- epilogue: O *= rcp(l), write bf16 [B*T, D] ----
  float inv_l[8];
#pragma unroll
  for (int r = 0; r < 8; ++r) inv_l[r] = __builtin_amdgcn_rcpf(l_i[r]);
#pragma unroll
  for (int di = 0; di < 4; ++di)
#pragma unroll
    for (int r = 0; r < 8; ++r) {
      const int row = qbase + w * 16 + r + 8 * g16;
      const int col = h * HD + di * 16 + ln;
      O[(size_t)(b * T + row) * D + col] = (__bf16)(ofrag[di][r] * inv_l[r]);
    }
}

// ---------------------------------------------------------------------------
// Host launcher
// ---------------------------------------------------------------------------
extern "C" void kernel_launch(void* const* d_in, const int* in_sizes, int n_in,
                              void* d_out, int out_size, void* d_ws,
                              size_t ws_size, hipStream_t stream) {
  constexpr int Bc = 2, T = 2048, Dd = 1024, Hh = 16, FFc = 4096;
  constexpr int Mr = Bc * T;  // 4096 rows

  const float* x      = (const float*)d_in[0];
  // d_in[1] = causal_mask (unused; causality computed analytically)
  const float* Wq     = (const float*)d_in[2];
  const float* Wk     = (const float*)d_in[3];
  const float* Wv     = (const float*)d_in[4];
  const float* Wo     = (const float*)d_in[5];
  const float* bo     = (const float*)d_in[6];
  const float* W1     = (const float*)d_in[7];
  const float* b1     = (const float*)d_in[8];
  const float* W2     = (const float*)d_in[9];
  const float* b2     = (const float*)d_in[10];
  const float* g_attn = (const float*)d_in[11];
  const float* g_ff   = (const float*)d_in[12];
  float* out = (float*)d_out;

  // ---- workspace carve-out ----
  char* ws = (char*)d_ws;
  size_t off = 0;
  auto alloc = [&](size_t bytes) -> char* {
    char* p = ws + off;
    off += (bytes + 255) / 256 * 256;
    return p;
  };
  __bf16* xa   = (__bf16*)alloc((size_t)Mr * Dd * 2);        // rmsnorm(x)
  __bf16* qb   = (__bf16*)alloc((size_t)Mr * Dd * 2);        // q (pre-scaled)
  __bf16* kb   = (__bf16*)alloc((size_t)Mr * Dd * 2);        // k
  __bf16* vT   = (__bf16*)alloc((size_t)Mr * Dd * 2);        // v, [B][H][HD][T]
  __bf16* ab   = (__bf16*)alloc((size_t)Mr * Dd * 2);        // attention out
  __bf16* xfb  = (__bf16*)alloc((size_t)Mr * Dd * 2);        // rmsnorm(x1)
  __bf16* hb   = (__bf16*)alloc((size_t)Mr * FFc * 2);       // silu(ff1)
  float*  x1   = (float*)alloc((size_t)Mr * Dd * 4);         // x + attn@Wo + bo
  __bf16* wqkv = (__bf16*)alloc((size_t)3 * Dd * Dd * 2);    // fused, transposed
  __bf16* wob  = (__bf16*)alloc((size_t)Dd * Dd * 2);
  __bf16* w1b  = (__bf16*)alloc((size_t)Dd * FFc * 2);
  __bf16* w2b  = (__bf16*)alloc((size_t)FFc * Dd * 2);
  (void)ws_size; (void)in_sizes; (void)n_in; (void)out_size;

  // ---- transpose+convert weights once: dst[N][K] = bf16(W[K][N]) ----
  auto tr = [&](const float* src, __bf16* dst, int K, int N) {
    dim3 g(N / 32, K / 32);
    transpose_to_bf16_kernel<<<g, 256, 0, stream>>>(src, dst, K, N);
  };
  tr(Wq, wqkv,                       Dd, Dd);
  tr(Wk, wqkv + (size_t)Dd * Dd,     Dd, Dd);
  tr(Wv, wqkv + (size_t)2 * Dd * Dd, Dd, Dd);
  tr(Wo, wob, Dd, Dd);
  tr(W1, w1b, Dd, FFc);
  tr(W2, w2b, FFc, Dd);

  // xa = rmsnorm(x, g_attn)
  rmsnorm_bf16_kernel<<<Mr, 256, 0, stream>>>(x, g_attn, xa, Dd);

  // fused qkv = xa @ [Wq|Wk|Wv]; q pre-scaled, v per-head-transposed
  gemm_bf16_wmma_kernel<false, false, false, OUT_QKV>
      <<<dim3(3 * Dd / 128, Mr / 128), 128, 0, stream>>>(
          xa, wqkv, nullptr, nullptr, nullptr, nullptr, qb, kb, vT,
          Mr, 3 * Dd, Dd);

  // causal flash attention per (b, h, 128-row q tile)
  flash_attn_bf16_kernel<<<Bc * Hh * (T / 128), 256, 0, stream>>>(qb, kb, vT, ab);

  // x1 = x + attn @ Wo + bo
  gemm_bf16_wmma_kernel<true, false, true, OUT_F32>
      <<<dim3(Dd / 128, Mr / 128), 128, 0, stream>>>(
          ab, wob, bo, x, x1, nullptr, nullptr, nullptr, nullptr, Mr, Dd, Dd);

  // xf = rmsnorm(x1, g_ff)
  rmsnorm_bf16_kernel<<<Mr, 256, 0, stream>>>(x1, g_ff, xfb, Dd);

  // h = silu(xf @ W1 + b1)
  gemm_bf16_wmma_kernel<true, true, false, OUT_BF16>
      <<<dim3(FFc / 128, Mr / 128), 128, 0, stream>>>(
          xfb, w1b, b1, nullptr, nullptr, hb, nullptr, nullptr, nullptr,
          Mr, FFc, Dd);

  // out = x1 + h @ W2 + b2
  gemm_bf16_wmma_kernel<true, false, true, OUT_F32>
      <<<dim3(Dd / 128, Mr / 128), 128, 0, stream>>>(
          hb, w2b, b2, x1, out, nullptr, nullptr, nullptr, nullptr,
          Mr, Dd, FFc);
}
